// MLP3A_91044716740669
// MI455X (gfx1250) — compile-verified, matrix operand order checked by reading, not run on hardware
//
#include <hip/hip_runtime.h>
#include <hip/hip_bf16.h>

// ---------------------------------------------------------------------------
// Shapes from the reference: B=1024, DIM=512, L=512 (DIM == L).
// Phase 1: qkv[b, 3l+c] = dot(x[b,:], qkv_w[3l+c,:])  -> q/k/v planes in d_ws,
//          in_proj scale+bias fused.  v_wmma_f32_16x16x32_bf16, with the A
//          (x) fragment reused across the 3 q/k/v components (3 WMMAs per
//          K-chunk per wave, 48 WMMAs per wave total).
// Phase 2: rank-1 attention per batch: softmax_j(q_i * k_j) @ v with analytic
//          row-max, K/V staged into LDS via the CDNA5 async-to-LDS path
//          (global_load_async_to_lds_b128 + s_wait_asynccnt), then out_proj
//          scalar affine + residual x.
// ---------------------------------------------------------------------------

#define BATCH 1024
#define DIM   512
#define SEQL  512

typedef __attribute__((ext_vector_type(16))) __bf16 v16bf;
typedef __attribute__((ext_vector_type(8)))  float  v8f;
typedef __attribute__((ext_vector_type(4)))  float  f32x4;

// ---------------------------------------------------------------------------
// Kernel 1: QKV projection GEMM via WMMA.
// One wave computes the 16x16 tile (16 batch rows x 16 seq cols) for ALL
// THREE components c in {0(q),1(k),2(v)}: W rows 3l, 3l+1, 3l+2 are
// contiguous, and the A fragment (x tile) is shared across the 3 WMMAs.
//
// ISA lane layouts (CDNA5 ISA §7.12.2), half = lane>>4, r = lane&15:
//   A (16x32 bf16): lane row M=r; a[i] = A[M][ (i/8)*16 + half*8 + (i%8) ]
//   B (32x16 bf16): lane col N=r; b[i] = B[ (i/8)*16 + half*8 + (i%8) ][N]
//   C/D (16x16 f32): c[j] = D[ j + 8*half ][ r ]
// ---------------------------------------------------------------------------
__global__ __launch_bounds__(256)
void qkv_wmma_kernel(const float* __restrict__ x,      // [B, DIM]
                     const float* __restrict__ w,      // [3*L, DIM]
                     const float* __restrict__ ipw,    // [3,1] flat
                     const float* __restrict__ ipb,    // [3]
                     float* __restrict__ qkv_ws)       // 3 planes of [B, L]
{
    const int lane = threadIdx.x & 31;
    const int wave = blockIdx.x * (blockDim.x >> 5) + (threadIdx.x >> 5);
    // 2048 tile-triples total = 64 b-tiles * 32 l-tiles
    const int ltile = wave & 31;
    const int btile = wave >> 5;

    const int half = lane >> 4;
    const int r    = lane & 15;

    const float* rowA  = x + (size_t)(btile * 16 + r) * DIM;   // A row M=r
    const float* rowB0 = w + (size_t)(3 * (ltile * 16 + r)) * DIM; // c=0 col N=r
    const float* rowB1 = rowB0 + DIM;                              // c=1
    const float* rowB2 = rowB0 + 2 * DIM;                          // c=2

    v8f acc0 = {}, acc1 = {}, acc2 = {};
#pragma unroll
    for (int k0 = 0; k0 < DIM; k0 += 32) {
        const int o0 = k0 + half * 8;        // K = o0 + 0..7
        const int o1 = o0 + 16;              // K = o1 + 0..7

        f32x4 a00 = *(const f32x4*)(rowA + o0);
        f32x4 a01 = *(const f32x4*)(rowA + o0 + 4);
        f32x4 a10 = *(const f32x4*)(rowA + o1);
        f32x4 a11 = *(const f32x4*)(rowA + o1 + 4);

        v16bf av;
#pragma unroll
        for (int t = 0; t < 4; ++t) {
            av[t]      = (__bf16)a00[t];
            av[t + 4]  = (__bf16)a01[t];
            av[t + 8]  = (__bf16)a10[t];
            av[t + 12] = (__bf16)a11[t];
        }

        const float* rb[3] = { rowB0, rowB1, rowB2 };
        v8f* accs[3] = { &acc0, &acc1, &acc2 };
#pragma unroll
        for (int c = 0; c < 3; ++c) {
            f32x4 b00 = *(const f32x4*)(rb[c] + o0);
            f32x4 b01 = *(const f32x4*)(rb[c] + o0 + 4);
            f32x4 b10 = *(const f32x4*)(rb[c] + o1);
            f32x4 b11 = *(const f32x4*)(rb[c] + o1 + 4);
            v16bf bv;
#pragma unroll
            for (int t = 0; t < 4; ++t) {
                bv[t]      = (__bf16)b00[t];
                bv[t + 4]  = (__bf16)b01[t];
                bv[t + 8]  = (__bf16)b10[t];
                bv[t + 12] = (__bf16)b11[t];
            }
            // (neg_a, A, neg_b, B, c_mod, C, reuse_a, reuse_b)
            *accs[c] = __builtin_amdgcn_wmma_f32_16x16x32_bf16(
                false, av, false, bv, (short)0, *accs[c], false, false);
        }
    }

    // Fused MHA in-projection (scalar affine per component), write planes.
    v8f* accs[3] = { &acc0, &acc1, &acc2 };
#pragma unroll
    for (int c = 0; c < 3; ++c) {
        const float wc = ipw[c];
        const float bc = ipb[c];
        float* plane = qkv_ws + (size_t)c * (BATCH * SEQL);
        const int l = ltile * 16 + r;
#pragma unroll
        for (int j = 0; j < 8; ++j) {
            const int bidx = btile * 16 + j + 8 * half;   // D row -> batch idx
            plane[(size_t)bidx * SEQL + l] = (*accs[c])[j] * wc + bc;
        }
    }
}

// ---------------------------------------------------------------------------
// Kernel 2: rank-1 attention, one workgroup per batch element.
// scores[i,j] = q_i * k_j  =>  row max is q_i * (q_i>=0 ? kmax : kmin).
// K/V rows (4 KB) staged into LDS with global_load_async_to_lds_b128:
// 256 threads x 16 B = 4 KB in a single async op per wave, completed with
// s_wait_asynccnt 0 before the workgroup barrier.
// ---------------------------------------------------------------------------
__global__ __launch_bounds__(256)
void rank1_attn_kernel(const float* __restrict__ qkv_ws, // q/k/v planes [B,L]
                       const float* __restrict__ x,      // [B, DIM] residual
                       const float* __restrict__ ow,     // [1,1]
                       const float* __restrict__ ob,     // [1]
                       float* __restrict__ out)          // [B, L]
{
    __shared__ float k_s[SEQL];
    __shared__ float v_s[SEQL];

    const int b   = blockIdx.x;
    const int tid = threadIdx.x;

    const float* qs = qkv_ws;
    const float* ks = qkv_ws + (size_t)1 * BATCH * SEQL;
    const float* vs = qkv_ws + (size_t)2 * BATCH * SEQL;

    // Async stage: threads 0-127 move the K row, 128-255 the V row, 16 B each.
    {
        const float* srow  = (tid < 128) ? (ks + (size_t)b * SEQL)
                                         : (vs + (size_t)b * SEQL);
        float*       dbase = (tid < 128) ? k_s : v_s;
        const int    e     = (tid & 127) * 4;   // 4 floats = 16 B per lane
        unsigned long long ga = (unsigned long long)(size_t)(srow + e);
        unsigned int       la = (unsigned int)(size_t)(dbase + e); // LDS offset
        asm volatile("global_load_async_to_lds_b128 %0, %1, off"
                     :: "v"(la), "v"(ga) : "memory");
        asm volatile("s_wait_asynccnt 0x0" ::: "memory");
    }
    __syncthreads();

    // Per-thread scan for kmax/kmin (512 LDS reads; trivial vs. exp cost).
    float kmax = -3.402823466e+38f, kmin = 3.402823466e+38f;
#pragma unroll 8
    for (int j = 0; j < SEQL; ++j) {
        const float kv = k_s[j];
        kmax = fmaxf(kmax, kv);
        kmin = fminf(kmin, kv);
    }

    const int i0 = tid;         // query row 0
    const int i1 = tid + 256;   // query row 1
    const float q0 = qs[(size_t)b * SEQL + i0];
    const float q1 = qs[(size_t)b * SEQL + i1];
    const float m0 = q0 >= 0.0f ? q0 * kmax : q0 * kmin;
    const float m1 = q1 >= 0.0f ? q1 * kmax : q1 * kmin;

    float se0 = 0.0f, sv0 = 0.0f;
    float se1 = 0.0f, sv1 = 0.0f;
#pragma unroll 4
    for (int j = 0; j < SEQL; ++j) {
        const float kv = k_s[j];
        const float vv = v_s[j];
        const float e0 = __expf(q0 * kv - m0);
        const float e1 = __expf(q1 * kv - m1);
        se0 += e0; sv0 += e0 * vv;
        se1 += e1; sv1 += e1 * vv;
    }

    const float owv = ow[0];
    const float obv = ob[0];
    out[(size_t)b * SEQL + i0] = (sv0 / se0) * owv + obv + x[(size_t)b * SEQL + i0];
    out[(size_t)b * SEQL + i1] = (sv1 / se1) * owv + obv + x[(size_t)b * SEQL + i1];
}

// ---------------------------------------------------------------------------
// Launch
// ---------------------------------------------------------------------------
extern "C" void kernel_launch(void* const* d_in, const int* in_sizes, int n_in,
                              void* d_out, int out_size, void* d_ws, size_t ws_size,
                              hipStream_t stream) {
    const float* x     = (const float*)d_in[0];  // [1024, 512]
    const float* qkv_w = (const float*)d_in[1];  // [1536, 512]
    const float* ipw   = (const float*)d_in[2];  // [3,1]
    const float* ipb   = (const float*)d_in[3];  // [3]
    const float* ow    = (const float*)d_in[4];  // [1,1]
    const float* ob    = (const float*)d_in[5];  // [1]
    float* out = (float*)d_out;                  // [1024, 512]

    float* qkv_ws = (float*)d_ws;                // 3 * 1024 * 512 floats = 6 MB

    // Phase 1: 2048 tile-triples, 8 waves (256 threads) per block -> 256 blocks.
    qkv_wmma_kernel<<<dim3(256), dim3(256), 0, stream>>>(x, qkv_w, ipw, ipb, qkv_ws);

    // Phase 2: one block per batch element.
    rank1_attn_kernel<<<dim3(BATCH), dim3(256), 0, stream>>>(qkv_ws, x, ow, ob, out);

    (void)in_sizes; (void)n_in; (void)out_size; (void)ws_size;
}
